// MultiHead_31722628448419
// MI455X (gfx1250) — matrix operand order repaired and checked by reading, not compiled, not run on hardware
//
#include <hip/hip_runtime.h>

#define B_ 4
#define S_ 2048
#define E_ 1024
#define H_ 16
#define D_ 64
#define M_ (B_ * S_)  // 8192 rows total

typedef __attribute__((ext_vector_type(16))) __bf16 v16bf;
typedef __attribute__((ext_vector_type(8)))  float  v8f;
typedef __attribute__((ext_vector_type(4)))  unsigned int u32x4;

union FragB16 {
  v16bf v;
  u32x4 q[2];
};

__device__ __forceinline__ unsigned short f2bf(float f) {
  unsigned u = __float_as_uint(f);
  u += 0x7FFFu + ((u >> 16) & 1u);  // round-to-nearest-even
  return (unsigned short)(u >> 16);
}

// A fragment (16x32 bf16): lane m=lane%16, half=lane/16; per lane two
// contiguous runs of 8 bf16 at [half*8, +8) and [16+half*8, +8).
__device__ __forceinline__ v16bf load_a_frag(const unsigned short* p, int half) {
  FragB16 f;
  f.q[0] = *reinterpret_cast<const u32x4*>(p + half * 8);
  f.q[1] = *reinterpret_cast<const u32x4*>(p + 16 + half * 8);
  return f.v;
}

// B fragment (32x16 bf16): lane n=lane%16 holds column n; same K order as A:
// two runs of 8 bf16 at [half*8, +8) and [16+half*8, +8) of the operand row.
__device__ __forceinline__ v16bf load_b_frag(const unsigned short* p, int half) {
  FragB16 f;
  f.q[0] = *reinterpret_cast<const u32x4*>(p + half * 8);
  f.q[1] = *reinterpret_cast<const u32x4*>(p + 16 + half * 8);
  return f.v;
}

__device__ __forceinline__ v8f wmma_bf16(v16bf a, v16bf b, v8f c) {
  return __builtin_amdgcn_wmma_f32_16x16x32_bf16(false, a, false, b, (short)0, c,
                                                 false, false);
}

// CDNA5 async copy global -> LDS (per-lane 16B), tracked by ASYNCcnt.
__device__ __forceinline__ void async_b128(unsigned lds_byte_off, const void* g) {
  asm volatile("global_load_async_to_lds_b128 %0, %1, off"
               :: "v"(lds_byte_off), "v"(g)
               : "memory");
}
#define S_WAIT_ASYNC(N) asm volatile("s_wait_asynccnt " #N ::: "memory")

// ---------------- conversion kernels ----------------

__global__ __launch_bounds__(256) void cvt_bf16_kernel(const float* __restrict__ src,
                                                       unsigned short* __restrict__ dst,
                                                       int n) {
  int i = blockIdx.x * 256 + threadIdx.x;
  int stride = gridDim.x * 256;
  for (; i < n; i += stride) dst[i] = f2bf(src[i]);
}

// src: nmat matrices of [rows, cols] f32 -> dst: nmat matrices of [cols, rows] bf16
__global__ __launch_bounds__(256) void transpose_cvt_kernel(const float* __restrict__ src,
                                                            unsigned short* __restrict__ dst,
                                                            int rows, int cols, int nmat) {
  int i = blockIdx.x * 256 + threadIdx.x;
  int stride = gridDim.x * 256;
  int per = rows * cols;
  int n = per * nmat;
  for (; i < n; i += stride) {
    int m = i / per;
    int rem = i - m * per;
    int r = rem / cols;
    int c = rem - r * cols;
    dst[(size_t)m * per + (size_t)c * rows + r] = f2bf(src[i]);
  }
}

// ---------------- QKV projection: one wave computes a 32x64 tile ----------------
__global__ __launch_bounds__(32) void qkv_proj_kernel(
    const unsigned short* __restrict__ Xbf,
    const unsigned short* __restrict__ WqT,
    const unsigned short* __restrict__ WkT,
    const unsigned short* __restrict__ WvT,
    unsigned short* __restrict__ Qb,
    unsigned short* __restrict__ Kb,
    unsigned short* __restrict__ Vt) {
  const int lane = threadIdx.x;
  const int n = lane & 15, half = lane >> 4;
  const int m0 = blockIdx.x * 32;
  const int h = blockIdx.y;  // head == 64-col tile
  const int z = blockIdx.z;  // 0=Q 1=K 2=V
  const unsigned short* WT = (z == 0) ? WqT : (z == 1) ? WkT : WvT;

  const unsigned short* arow0 = Xbf + (size_t)(m0 + n) * E_;
  const unsigned short* arow1 = Xbf + (size_t)(m0 + 16 + n) * E_;
  const unsigned short* br[4];
#pragma unroll
  for (int dt = 0; dt < 4; ++dt)
    br[dt] = WT + ((size_t)h * D_ + dt * 16 + n) * E_;

  v8f acc[2][4] = {};
  for (int k0 = 0; k0 < E_; k0 += 32) {
    __builtin_prefetch(arow0 + k0 + 512);
    __builtin_prefetch(br[0] + k0 + 512);
    // hoist all fragment loads so the scheduler can rotate buffers
    v16bf a0 = load_a_frag(arow0 + k0, half);
    v16bf a1 = load_a_frag(arow1 + k0, half);
    v16bf b0 = load_b_frag(br[0] + k0, half);
    v16bf b1 = load_b_frag(br[1] + k0, half);
    v16bf b2 = load_b_frag(br[2] + k0, half);
    v16bf b3 = load_b_frag(br[3] + k0, half);
    acc[0][0] = wmma_bf16(a0, b0, acc[0][0]);
    acc[1][0] = wmma_bf16(a1, b0, acc[1][0]);
    acc[0][1] = wmma_bf16(a0, b1, acc[0][1]);
    acc[1][1] = wmma_bf16(a1, b1, acc[1][1]);
    acc[0][2] = wmma_bf16(a0, b2, acc[0][2]);
    acc[1][2] = wmma_bf16(a1, b2, acc[1][2]);
    acc[0][3] = wmma_bf16(a0, b3, acc[0][3]);
    acc[1][3] = wmma_bf16(a1, b3, acc[1][3]);
  }

  const int bidx = m0 / S_;
  const int s0 = m0 % S_;
#pragma unroll
  for (int u = 0; u < 2; ++u) {
#pragma unroll
    for (int dt = 0; dt < 4; ++dt) {
      const int dcol = dt * 16 + n;
#pragma unroll
      for (int r = 0; r < 8; ++r) {
        int srow = s0 + u * 16 + r + half * 8;
        unsigned short v = f2bf(acc[u][dt][r]);
        if (z == 2) {
          Vt[(((size_t)bidx * H_ + h) * D_ + dcol) * S_ + srow] = v;
        } else {
          unsigned short* dst = (z == 0) ? Qb : Kb;
          dst[(((size_t)bidx * H_ + h) * S_ + srow) * D_ + dcol] = v;
        }
      }
    }
  }
}

// ---------------- flash attention ----------------
// 8 waves / workgroup; WG covers 256 queries of one (b,h). K/V 32-key tiles are
// double-buffered in LDS via global_load_async_to_lds_b128 (ASYNCcnt), shared
// by all 8 waves. Each wave owns 32 queries.
__global__ __launch_bounds__(256) void attn_kernel(
    const unsigned short* __restrict__ Qb,   // [B,H,S,D]
    const unsigned short* __restrict__ Kb,   // [B,H,S,D]
    const unsigned short* __restrict__ Vt,   // [B,H,D,S]
    unsigned short* __restrict__ Ob) {       // [B,S,H*D]
  __shared__ __align__(16) unsigned short Klds[2][32 * 64];  // 2 x 4 KB [key][d]
  __shared__ __align__(16) unsigned short Vlds[2][64 * 32];  // 2 x 4 KB [d][key]
  __shared__ __align__(16) unsigned short Pl[8][32 * 32];    // 8 x 2 KB per-wave P

  const int tid = threadIdx.x;
  const int lane = tid & 31;
  const int w = tid >> 5;
  const int n = lane & 15, half = lane >> 4;
  const int wgq0 = blockIdx.x * 256;
  const int q0 = wgq0 + w * 32;  // this wave's 32 queries
  const int bh = blockIdx.y;
  const unsigned short* Qh = Qb + (size_t)bh * S_ * D_;
  const unsigned short* Kh = Kb + (size_t)bh * S_ * D_;
  const unsigned short* Vh = Vt + (size_t)bh * D_ * S_;

  const unsigned kbase[2] = {(unsigned)(size_t)&Klds[0][0],
                             (unsigned)(size_t)&Klds[1][0]};
  const unsigned vbase[2] = {(unsigned)(size_t)&Vlds[0][0],
                             (unsigned)(size_t)&Vlds[1][0]};
  const unsigned g = (unsigned)(w * 32 + lane);  // 0..255: each thread moves 16B

  // Q fragments: 2 query tiles x 2 k-steps, loaded once
  v16bf qf[2][2];
#pragma unroll
  for (int u = 0; u < 2; ++u) {
    const unsigned short* qrow = Qh + (size_t)(q0 + u * 16 + n) * D_;
    qf[u][0] = load_a_frag(qrow, half);
    qf[u][1] = load_a_frag(qrow + 32, half);
  }

  v8f o[2][4] = {};
  float mst[2][8], lst[2][8];
#pragma unroll
  for (int u = 0; u < 2; ++u)
#pragma unroll
    for (int r = 0; r < 8; ++r) { mst[u][r] = -1e30f; lst[u][r] = 0.f; }
  const float scale = 0.125f;  // D^-0.5

  // stage one 32-key K/V tile pair into buffer `buf` (2 async ops per wave)
  auto stage = [&](int buf, int n0) {
    // K tile: 32 rows x 128B == 4KB contiguous in [S,D] layout
    const char* gk = (const char*)(Kh + (size_t)n0 * D_) + g * 16u;
    async_b128(kbase[buf] + g * 16u, gk);
    // V tile: 64 rows (d) x 64B, rows strided by S*2 bytes
    int d = g >> 2;
    const char* gv = (const char*)(Vh + (size_t)d * S_ + n0) + (g & 3u) * 16u;
    async_b128(vbase[buf] + g * 16u, gv);
  };

  const int it_total = (wgq0 + 256) / 32;  // causal bound for whole WG
  stage(0, 0);
  for (int it = 0; it < it_total; ++it) {
    const int n0 = it * 32;
    const int cur = it & 1;
    const bool has_next = (it + 1) < it_total;
    if (has_next) {
      stage(cur ^ 1, n0 + 32);
      S_WAIT_ASYNC(2);  // current buffer's 2 ops complete (in-order)
    } else {
      S_WAIT_ASYNC(0);
    }
    __syncthreads();  // all waves' slices of `cur` are in LDS

    if (n0 < q0 + 32) {  // uniform per wave: causal work remains
      const unsigned short* Kc = Klds[cur];
      const unsigned short* Vc = Vlds[cur];
      v16bf kb[2][2];
#pragma unroll
      for (int t = 0; t < 2; ++t) {
        kb[t][0] = load_b_frag(Kc + (size_t)(t * 16 + n) * 64, half);
        kb[t][1] = load_b_frag(Kc + (size_t)(t * 16 + n) * 64 + 32, half);
      }
      v16bf vb[4];
#pragma unroll
      for (int dt = 0; dt < 4; ++dt)
        vb[dt] = load_b_frag(Vc + (size_t)(dt * 16 + n) * 32, half);

      v8f sc[2][2];
#pragma unroll
      for (int u = 0; u < 2; ++u) {
#pragma unroll
        for (int t = 0; t < 2; ++t) {
          v8f s = {};
          s = wmma_bf16(qf[u][0], kb[t][0], s);
          s = wmma_bf16(qf[u][1], kb[t][1], s);
          sc[u][t] = s;
        }
      }
#pragma unroll
      for (int u = 0; u < 2; ++u) {
#pragma unroll
        for (int t = 0; t < 2; ++t) {
#pragma unroll
          for (int r = 0; r < 8; ++r) {
            int q = q0 + u * 16 + r + half * 8;
            int key = n0 + t * 16 + n;
            float v = sc[u][t][r] * scale;
            sc[u][t][r] = (key <= q) ? v : -1e30f;
          }
        }
#pragma unroll
        for (int r = 0; r < 8; ++r) {
          float rm = fmaxf(sc[u][0][r], sc[u][1][r]);
          rm = fmaxf(rm, __shfl_xor(rm, 1, 32));
          rm = fmaxf(rm, __shfl_xor(rm, 2, 32));
          rm = fmaxf(rm, __shfl_xor(rm, 4, 32));
          rm = fmaxf(rm, __shfl_xor(rm, 8, 32));
          float mnew = fmaxf(mst[u][r], rm);
          float corr = __expf(mst[u][r] - mnew);
          mst[u][r] = mnew;
          float p0 = __expf(sc[u][0][r] - mnew);
          float p1 = __expf(sc[u][1][r] - mnew);
          sc[u][0][r] = p0;
          sc[u][1][r] = p1;
          float rs = p0 + p1;
          rs += __shfl_xor(rs, 1, 32);
          rs += __shfl_xor(rs, 2, 32);
          rs += __shfl_xor(rs, 4, 32);
          rs += __shfl_xor(rs, 8, 32);
          lst[u][r] = lst[u][r] * corr + rs;
#pragma unroll
          for (int dt = 0; dt < 4; ++dt) o[u][dt][r] *= corr;
        }
      }
      // C-layout -> A-layout for P via this wave's private LDS region
      unsigned short* Pw = Pl[w];
#pragma unroll
      for (int u = 0; u < 2; ++u)
#pragma unroll
        for (int t = 0; t < 2; ++t)
#pragma unroll
          for (int r = 0; r < 8; ++r)
            Pw[(u * 16 + r + half * 8) * 32 + t * 16 + n] = f2bf(sc[u][t][r]);
      v16bf pf0 = load_a_frag(Pw + (size_t)n * 32, half);
      v16bf pf1 = load_a_frag(Pw + (size_t)(16 + n) * 32, half);
#pragma unroll
      for (int dt = 0; dt < 4; ++dt) {
        o[0][dt] = wmma_bf16(pf0, vb[dt], o[0][dt]);
        o[1][dt] = wmma_bf16(pf1, vb[dt], o[1][dt]);
      }
    }
    __syncthreads();  // everyone done reading `cur` before it is restaged
  }

  const int b = bh / H_, h = bh % H_;
#pragma unroll
  for (int u = 0; u < 2; ++u) {
#pragma unroll
    for (int r = 0; r < 8; ++r) {
      float inv = 1.0f / lst[u][r];
      size_t row = (size_t)(b * S_ + q0 + u * 16 + r + half * 8) * (H_ * D_) +
                   (size_t)h * D_ + n;
      Ob[row + 0]  = f2bf(o[u][0][r] * inv);
      Ob[row + 16] = f2bf(o[u][1][r] * inv);
      Ob[row + 32] = f2bf(o[u][2][r] * inv);
      Ob[row + 48] = f2bf(o[u][3][r] * inv);
    }
  }
}

// ---------------- output projection + bias: one wave computes 32x64 ----------------
__global__ __launch_bounds__(32) void oproj_kernel(
    const unsigned short* __restrict__ Ob,
    const unsigned short* __restrict__ WoT,
    const float* __restrict__ bo,
    float* __restrict__ out) {
  const int lane = threadIdx.x;
  const int n = lane & 15, half = lane >> 4;
  const int m0 = blockIdx.x * 32;
  const int e0 = blockIdx.y * 64;
  const unsigned short* arow0 = Ob + (size_t)(m0 + n) * (H_ * D_);
  const unsigned short* arow1 = Ob + (size_t)(m0 + 16 + n) * (H_ * D_);
  const unsigned short* br[4];
#pragma unroll
  for (int et = 0; et < 4; ++et)
    br[et] = WoT + (size_t)(e0 + et * 16 + n) * (H_ * D_);

  v8f acc[2][4] = {};
  for (int k0 = 0; k0 < H_ * D_; k0 += 32) {
    __builtin_prefetch(arow0 + k0 + 512);
    __builtin_prefetch(br[0] + k0 + 512);
    v16bf a0 = load_a_frag(arow0 + k0, half);
    v16bf a1 = load_a_frag(arow1 + k0, half);
    v16bf b0 = load_b_frag(br[0] + k0, half);
    v16bf b1 = load_b_frag(br[1] + k0, half);
    v16bf b2 = load_b_frag(br[2] + k0, half);
    v16bf b3 = load_b_frag(br[3] + k0, half);
    acc[0][0] = wmma_bf16(a0, b0, acc[0][0]);
    acc[1][0] = wmma_bf16(a1, b0, acc[1][0]);
    acc[0][1] = wmma_bf16(a0, b1, acc[0][1]);
    acc[1][1] = wmma_bf16(a1, b1, acc[1][1]);
    acc[0][2] = wmma_bf16(a0, b2, acc[0][2]);
    acc[1][2] = wmma_bf16(a1, b2, acc[1][2]);
    acc[0][3] = wmma_bf16(a0, b3, acc[0][3]);
    acc[1][3] = wmma_bf16(a1, b3, acc[1][3]);
  }
#pragma unroll
  for (int et = 0; et < 4; ++et) {
    const float bias = bo[e0 + et * 16 + n];
#pragma unroll
    for (int u = 0; u < 2; ++u) {
#pragma unroll
      for (int r = 0; r < 8; ++r) {
        int m = m0 + u * 16 + r + half * 8;
        out[(size_t)m * E_ + e0 + et * 16 + n] = acc[u][et][r] + bias;
      }
    }
  }
}

extern "C" void kernel_launch(void* const* d_in, const int* in_sizes, int n_in,
                              void* d_out, int out_size, void* d_ws, size_t ws_size,
                              hipStream_t stream) {
  (void)in_sizes; (void)n_in; (void)out_size; (void)ws_size;
  const float* X  = (const float*)d_in[0];
  const float* Wq = (const float*)d_in[1];
  const float* Wk = (const float*)d_in[2];
  const float* Wv = (const float*)d_in[3];
  const float* Wo = (const float*)d_in[4];
  const float* bo = (const float*)d_in[5];
  float* out = (float*)d_out;

  char* ws = (char*)d_ws;
  size_t off = 0;
  auto walloc = [&](size_t bytes) -> char* {
    char* p = ws + off;
    off += (bytes + 255) & ~(size_t)255;
    return p;
  };
  unsigned short* Xbf = (unsigned short*)walloc((size_t)M_ * E_ * 2);
  unsigned short* WqT = (unsigned short*)walloc((size_t)H_ * E_ * D_ * 2);
  unsigned short* WkT = (unsigned short*)walloc((size_t)H_ * E_ * D_ * 2);
  unsigned short* WvT = (unsigned short*)walloc((size_t)H_ * E_ * D_ * 2);
  unsigned short* WoT = (unsigned short*)walloc((size_t)H_ * D_ * E_ * 2);
  unsigned short* Qb  = (unsigned short*)walloc((size_t)B_ * H_ * S_ * D_ * 2);
  unsigned short* Kb  = (unsigned short*)walloc((size_t)B_ * H_ * S_ * D_ * 2);
  unsigned short* Vt  = (unsigned short*)walloc((size_t)B_ * H_ * D_ * S_ * 2);
  unsigned short* Ob  = (unsigned short*)walloc((size_t)M_ * H_ * D_ * 2);

  cvt_bf16_kernel<<<4096, 256, 0, stream>>>(X, Xbf, M_ * E_);
  transpose_cvt_kernel<<<2048, 256, 0, stream>>>(Wq, WqT, E_, D_, H_);
  transpose_cvt_kernel<<<2048, 256, 0, stream>>>(Wk, WkT, E_, D_, H_);
  transpose_cvt_kernel<<<2048, 256, 0, stream>>>(Wv, WvT, E_, D_, H_);
  transpose_cvt_kernel<<<2048, 256, 0, stream>>>(Wo, WoT, H_ * D_, E_, 1);

  qkv_proj_kernel<<<dim3(M_ / 32, H_, 3), 32, 0, stream>>>(
      Xbf, WqT, WkT, WvT, Qb, Kb, Vt);

  attn_kernel<<<dim3(S_ / 256, B_ * H_), 256, 0, stream>>>(Qb, Kb, Vt, Ob);

  oproj_kernel<<<dim3(M_ / 32, E_ / 64), 32, 0, stream>>>(Ob, WoT, bo, out);
}
